// PCSQCNN_75977971466575
// MI455X (gfx1250) — compile-verified, matrix operand order checked by reading, not run on hardware
//
#include <hip/hip_runtime.h>
#include <hip/hip_bf16.h>

#define PI_F 3.14159265358979f

typedef unsigned int u32x4 __attribute__((ext_vector_type(4)));
typedef int          i32x8 __attribute__((ext_vector_type(8)));
typedef int          i32x4 __attribute__((ext_vector_type(4)));
typedef _Float16     v8h   __attribute__((ext_vector_type(8)));
typedef _Float16     v16h  __attribute__((ext_vector_type(16)));
typedef float        v8f   __attribute__((ext_vector_type(8)));

// ---------------------------------------------------------------------------
// Prep kernel 1: build per-layer 2x2 unitary tables from Pauli coefficients,
// written at BIT-REVERSED (mx,my) so the fused kernel can read them linearly
// while its state sits in DIF (bit-reversed) Fourier order.
// Layout: T[cc][mx_rev][my_rev][8] = {u00.re,u00.im,u01.re,u01.im,u10.re,u10.im,u11.re,u11.im}
// ---------------------------------------------------------------------------
__device__ __forceinline__ int bitrev(int x, int nb) {
    int r = 0;
    for (int i = 0; i < nb; ++i) { r = (r << 1) | (x & 1); x >>= 1; }
    return r;
}

__global__ void build_mux_table(const float* __restrict__ C, float* __restrict__ T,
                                int cdim, int L, int logL) {
    int t = blockIdx.x * blockDim.x + threadIdx.x;
    int total = cdim * L * L;
    if (t >= total) return;
    int my = t % L;
    int mx = (t / L) % L;
    int cc = t / (L * L);
    const float* c = C + (size_t)t * 3;   // [cc][mx][my][3] flat == t*3
    float ax = c[0], ay = c[1], az = c[2];
    float r  = sqrtf(ax * ax + ay * ay + az * az + 1e-20f);
    float cr = cosf(r);
    float sn = sinf(r) / r;
    size_t dst = ((size_t)(cc * L + bitrev(mx, logL)) * L + bitrev(my, logL)) * 8;
    T[dst + 0] = cr;       T[dst + 1] = -az * sn;   // u00
    T[dst + 2] = -ay * sn; T[dst + 3] = -ax * sn;   // u01
    T[dst + 4] =  ay * sn; T[dst + 5] = -ax * sn;   // u10
    T[dst + 6] = cr;       T[dst + 7] =  az * sn;   // u11
}

// ---------------------------------------------------------------------------
// Prep kernel 2: W [10,2048] f32 -> Wp [16,2048] f16 (rows 10..15 zero).
// ---------------------------------------------------------------------------
__global__ void build_wpad(const float* __restrict__ W, _Float16* __restrict__ Wp) {
    int t = blockIdx.x * blockDim.x + threadIdx.x;
    if (t >= 16 * 2048) return;
    int n = t >> 11, k = t & 2047;
    Wp[t] = (n < 10) ? (_Float16)W[n * 2048 + k] : (_Float16)0.f;
}

// ---------------------------------------------------------------------------
// Fused per-image pipeline. One workgroup (1024 threads) per image; the full
// complex state [128][128][2] lives in 256KB of LDS as float2 (re,im).
// Flat complex index: c = (x*128 + y)*2 + f  ->  c = x*256 + y*2 + f.
// Forward FFT = DIF (natural in, bit-reversed out); inverse = DIT with 0.5
// per stage (ortho qft+iqft pair => exact 1/L).  Pool is free (relabeling).
// ---------------------------------------------------------------------------
__device__ __forceinline__ void fftX_fwd(float2* S, int logL, int t) {
    for (int ls = logL - 1; ls >= 0; --ls) {
        int s = 1 << ls;
        float wstep = -PI_F / (float)s;
        for (int i = 0; i < 16; ++i) {
            int q    = (i << 10) + t;
            int line = q & 255;              // y*2+f : consecutive lanes -> conflict-free
            int bq   = q >> 8;               // 0..63 butterfly along x
            int j    = bq & (s - 1);
            int x1   = ((bq >> ls) << (ls + 1)) | j;
            int c1   = (x1 << 8) + line, c2 = c1 + (s << 8);
            float2 u = S[c1], v = S[c2];
            float wr, wi; __sincosf(wstep * (float)j, &wi, &wr);
            float dx = u.x - v.x, dy = u.y - v.y;
            S[c1] = make_float2(u.x + v.x, u.y + v.y);
            S[c2] = make_float2(dx * wr - dy * wi, dx * wi + dy * wr);
        }
        __syncthreads();
    }
}

__device__ __forceinline__ void fftX_inv(float2* S, int logL, int t) {
    for (int ls = 0; ls < logL; ++ls) {
        int s = 1 << ls;
        float wstep = PI_F / (float)s;
        for (int i = 0; i < 16; ++i) {
            int q    = (i << 10) + t;
            int line = q & 255;
            int bq   = q >> 8;
            int j    = bq & (s - 1);
            int x1   = ((bq >> ls) << (ls + 1)) | j;
            int c1   = (x1 << 8) + line, c2 = c1 + (s << 8);
            float2 u = S[c1], v = S[c2];
            float wr, wi; __sincosf(wstep * (float)j, &wi, &wr);
            float tx = v.x * wr - v.y * wi, ty = v.x * wi + v.y * wr;
            S[c1] = make_float2(0.5f * (u.x + tx), 0.5f * (u.y + ty));
            S[c2] = make_float2(0.5f * (u.x - tx), 0.5f * (u.y - ty));
        }
        __syncthreads();
    }
}

__device__ __forceinline__ void fftY_fwd(float2* S, int logL, int t) {
    for (int ls = logL - 1; ls >= 0; --ls) {
        int s = 1 << ls;
        float wstep = -PI_F / (float)s;
        for (int i = 0; i < 16; ++i) {
            int q    = (i << 10) + t;
            int bq   = q & 63;               // butterfly along y (fast across lanes)
            int line = q >> 6;               // x*2+f
            int j    = bq & (s - 1);
            int y1   = ((bq >> ls) << (ls + 1)) | j;
            int c1   = ((line >> 1) << 8) + (y1 << 1) + (line & 1);
            int c2   = c1 + (s << 1);
            float2 u = S[c1], v = S[c2];
            float wr, wi; __sincosf(wstep * (float)j, &wi, &wr);
            float dx = u.x - v.x, dy = u.y - v.y;
            S[c1] = make_float2(u.x + v.x, u.y + v.y);
            S[c2] = make_float2(dx * wr - dy * wi, dx * wi + dy * wr);
        }
        __syncthreads();
    }
}

__device__ __forceinline__ void fftY_inv(float2* S, int logL, int t) {
    for (int ls = 0; ls < logL; ++ls) {
        int s = 1 << ls;
        float wstep = PI_F / (float)s;
        for (int i = 0; i < 16; ++i) {
            int q    = (i << 10) + t;
            int bq   = q & 63;
            int line = q >> 6;
            int j    = bq & (s - 1);
            int y1   = ((bq >> ls) << (ls + 1)) | j;
            int c1   = ((line >> 1) << 8) + (y1 << 1) + (line & 1);
            int c2   = c1 + (s << 1);
            float2 u = S[c1], v = S[c2];
            float wr, wi; __sincosf(wstep * (float)j, &wi, &wr);
            float tx = v.x * wr - v.y * wi, ty = v.x * wi + v.y * wr;
            S[c1] = make_float2(0.5f * (u.x + tx), 0.5f * (u.y + ty));
            S[c2] = make_float2(0.5f * (u.x - tx), 0.5f * (u.y - ty));
        }
        __syncthreads();
    }
}

// Apply 2x2 unitary per (bit-reversed) Fourier mode, conditioned on the most
// recently pooled bit per axis (LSB of the condition register = x>>logL & 1).
__device__ __forceinline__ void mux_apply(float2* S, const float* __restrict__ T,
                                          int logL, int cmask, int cyd, int t) {
    int L = 1 << logL;
    for (int i = 0; i < 16; ++i) {
        int q  = (i << 10) + t;          // site (x,y)
        int y  = q & 127, x = q >> 7;
        int xa = x & (L - 1), ya = y & (L - 1);
        int cc = ((x >> logL) & cmask) * cyd + ((y >> logL) & cmask);
        const float4* Tq = (const float4*)(T + (((size_t)(cc * L + xa) * L + ya) << 3));
        float4 uA = Tq[0];               // u00, u01
        float4 uB = Tq[1];               // u10, u11
        float2 p0 = S[2 * q], p1 = S[2 * q + 1];
        float2 o0, o1;
        o0.x = uA.x * p0.x - uA.y * p0.y + uA.z * p1.x - uA.w * p1.y;
        o0.y = uA.x * p0.y + uA.y * p0.x + uA.z * p1.y + uA.w * p1.x;
        o1.x = uB.x * p0.x - uB.y * p0.y + uB.z * p1.x - uB.w * p1.y;
        o1.y = uB.x * p0.y + uB.y * p0.x + uB.z * p1.y + uB.w * p1.x;
        S[2 * q]     = o0;
        S[2 * q + 1] = o1;
    }
    __syncthreads();
}

__global__ __launch_bounds__(1024, 1)
void pcs_forward(const float* __restrict__ images,
                 const float* __restrict__ T0, const float* __restrict__ T1,
                 const float* __restrict__ T2, _Float16* __restrict__ P) {
    extern __shared__ float2 S[];        // 32768 complex = 256KB
    float* Sf = (float*)S;
    const int t = threadIdx.x;
    const int b = blockIdx.x;

    // ---- TDM: DMA the 128x128 f32 image tile into LDS staging (float off 49152)
    if (t == 0) {
        unsigned long long ga = (unsigned long long)(const void*)(images + (size_t)b * 16384);
        unsigned lds_off = (unsigned)(unsigned long long)(Sf + 49152);
        u32x4 g0 = { 1u,                                  // count=1, user descriptor
                     lds_off,                             // lds_addr (bytes)
                     (unsigned)ga,                        // global_addr[31:0]
                     (unsigned)((ga >> 32) & 0x1FFFFFFull) | (2u << 30) }; // ga[56:32] | type=2
        i32x8 g1 = { 0x20000,                             // mask=0, data_size=2 (4B)
                     (128 << 16),                         // tensor_dim0 lo
                     (128 << 16),                         // tensor_dim1 lo
                     (128 << 16),                         // tile_dim0
                     128,                                 // tile_dim1 (tile_dim2=0)
                     128,                                 // tensor_dim0_stride lo
                     (16384 << 16),                       // tensor_dim1_stride lo
                     0 };
        i32x4 g2 = { 0, 0, 0, 0 };
        i32x4 g3 = { 0, 0, 0, 0 };
        i32x8 g4 = { 0, 0, 0, 0, 0, 0, 0, 0 };
        __builtin_amdgcn_tensor_load_to_lds(g0, g1, g2, g3, g4, 0);
        __builtin_amdgcn_s_wait_tensorcnt(0);
    }
    __syncthreads();

    // ---- FRQI encode: amp = [cos(pi*img/2), sin(pi*img/2)] / 128 (real), in place
    float v[16];
    for (int i = 0; i < 16; ++i) v[i] = Sf[49152 + (i << 10) + t];
    __syncthreads();
    for (int i = 0; i < 16; ++i) {
        int s = (i << 10) + t;
        float sn, cs; __sincosf(0.5f * PI_F * v[i], &sn, &cs);
        S[2 * s]     = make_float2(cs * 0.0078125f, 0.f);
        S[2 * s + 1] = make_float2(sn * 0.0078125f, 0.f);
    }
    __syncthreads();

    // ---- layer 0 : L=128, no conditioning
    fftX_fwd(S, 7, t); fftY_fwd(S, 7, t);
    mux_apply(S, T0, 7, 0, 1, t);
    fftY_inv(S, 7, t); fftX_inv(S, 7, t);
    // pool (free relabel) -> layer 1 : L=64, cx=cy=2
    fftX_fwd(S, 6, t); fftY_fwd(S, 6, t);
    mux_apply(S, T1, 6, 1, 2, t);
    fftY_inv(S, 6, t); fftX_inv(S, 6, t);
    // pool -> layer 2 : L=32, cx=cy=2
    fftX_fwd(S, 5, t); fftY_fwd(S, 5, t);
    mux_apply(S, T2, 5, 1, 2, t);
    fftY_inv(S, 5, t); fftX_inv(S, 5, t);

    // ---- marginal |.|^2 over condition dims (xc,yc in 0..3) -> P[b][2048] (f16)
    for (int r = 0; r < 2; ++r) {
        int o  = (r << 10) + t;          // (xa*32 + ya)*2 + f
        int f  = o & 1, ya = (o >> 1) & 31, xa = o >> 6;
        float acc = 0.f;
        for (int xc = 0; xc < 4; ++xc)
            for (int yc = 0; yc < 4; ++yc) {
                int c = (((xc * 32 + xa) << 7) + (yc * 32 + ya)) * 2 + f;
                float2 z = S[c];
                acc += z.x * z.x + z.y * z.y;
            }
        P[(size_t)b * 2048 + o] = (_Float16)acc;
    }
}

// ---------------------------------------------------------------------------
// Classifier GEMM: out[512,10] = P[512,2048] @ Wp^T[2048,16] + bias, via
// v_wmma_f32_16x16x32_f16. One wave per 16-row tile; K loop of 64 WMMAs.
// A: lane%16 = M, (lane/16)*8 = K-subgroup; halves [0..7]=K+kg, [8..15]=K+16+kg.
// B (column-major == rows of Wp): identical indexing with N = lane%16.
// D: v8f, element v -> row (v + 8*(lane/16)), col lane%16.
// ---------------------------------------------------------------------------
__global__ __launch_bounds__(32)
void classify(const _Float16* __restrict__ P, const _Float16* __restrict__ Wp,
              const float* __restrict__ bias, float* __restrict__ out) {
    int lane = threadIdx.x & 31;
    int m0   = blockIdx.x * 16;
    int rm   = lane & 15;
    int kg   = (lane >> 4) << 3;         // 0 or 8
    const _Float16* arow = P  + (size_t)(m0 + rm) * 2048;
    const _Float16* brow = Wp + (size_t)rm * 2048;
    v8f acc = {};
    for (int kt = 0; kt < 2048; kt += 32) {
        v8h a0 = *(const v8h*)(arow + kt + kg);
        v8h a1 = *(const v8h*)(arow + kt + 16 + kg);
        v8h b0 = *(const v8h*)(brow + kt + kg);
        v8h b1 = *(const v8h*)(brow + kt + 16 + kg);
        v16h a, bf;
        for (int i = 0; i < 8; ++i) { a[i] = a0[i]; a[i + 8] = a1[i];
                                      bf[i] = b0[i]; bf[i + 8] = b1[i]; }
        acc = __builtin_amdgcn_wmma_f32_16x16x32_f16(false, a, false, bf,
                                                     (short)0, acc, false, false);
    }
    int n = lane & 15;
    if (n < 10) {
        int mb = (lane >> 4) << 3;
        for (int v = 0; v < 8; ++v)
            out[(size_t)(m0 + mb + v) * 10 + n] = acc[v] + bias[n];
    }
}

// ---------------------------------------------------------------------------
// Launch
// ---------------------------------------------------------------------------
extern "C" void kernel_launch(void* const* d_in, const int* in_sizes, int n_in,
                              void* d_out, int out_size, void* d_ws, size_t ws_size,
                              hipStream_t stream) {
    const float* images = (const float*)d_in[0];
    const float* mux0   = (const float*)d_in[1];
    const float* mux1   = (const float*)d_in[2];
    const float* mux2   = (const float*)d_in[3];
    const float* W      = (const float*)d_in[4];
    const float* bias   = (const float*)d_in[5];
    float*       out    = (float*)d_out;

    char* ws = (char*)d_ws;
    float*    T0 = (float*)(ws);                   // 1*128*128*8 f32 = 512KB
    float*    T1 = (float*)(ws + 524288);          // 4* 64* 64*8 f32 = 512KB
    float*    T2 = (float*)(ws + 1048576);         // 4* 32* 32*8 f32 = 128KB
    _Float16* Wp = (_Float16*)(ws + 1179648);      // 16*2048 f16      =  64KB
    _Float16* P  = (_Float16*)(ws + 1245184);      // 512*2048 f16     =   2MB

    build_mux_table<<<(16384 + 255) / 256, 256, 0, stream>>>(mux0, T0, 1, 128, 7);
    build_mux_table<<<(16384 + 255) / 256, 256, 0, stream>>>(mux1, T1, 4, 64, 6);
    build_mux_table<<<(4096  + 255) / 256, 256, 0, stream>>>(mux2, T2, 4, 32, 5);
    build_wpad<<<(32768 + 255) / 256, 256, 0, stream>>>(W, Wp);

    // whole-image state (256KB) in LDS: request > default dynamic-LDS cap
    (void)hipFuncSetAttribute((const void*)pcs_forward,
                              hipFuncAttributeMaxDynamicSharedMemorySize, 262144);
    pcs_forward<<<512, 1024, 262144, stream>>>(images, T0, T1, T2, P);

    classify<<<32, 32, 0, stream>>>(P, Wp, bias, out);
}
